// StableMicrolensingTransformer_71305047049050
// MI455X (gfx1250) — matrix operand order repaired and checked by reading, not compiled, not run on hardware
//
#include <hip/hip_runtime.h>
#include <hip/hip_bf16.h>

typedef __bf16 bf16;
typedef __attribute__((ext_vector_type(16))) __bf16 v16bf;
typedef __attribute__((ext_vector_type(8)))  float  v8f;
typedef __attribute__((ext_vector_type(4)))  unsigned int u32x4;
typedef __attribute__((ext_vector_type(8)))  int i32x8;
typedef __attribute__((ext_vector_type(4)))  int i32x4;

#define B_   16
#define T_   1500
#define D_   128
#define H_   8
#define DK_  16
#define FF_  512
#define M_TOT (B_*T_)        // 24000, divisible by 64
#define TPAD 1504            // 47*32, padded key length
#define NQT  94              // ceil(1500/16)
#define NKT  47              // TPAD/32
#define LDSS 40              // LDS row stride (bf16 elems), 80B keeps 16B alignment

union Frag { uint4 u[2]; v16bf v; };

__device__ __forceinline__ float clampf(float x, float lo, float hi) {
    return fminf(fmaxf(x, lo), hi);
}
__device__ __forceinline__ float gelu_f(float x) {
    return 0.5f * x * (1.0f + erff(x * 0.70710678118654752f));
}

// ---------------- global safe-normalize stats + mask ----------------
__global__ __launch_bounds__(256) void stats_kernel(const float* __restrict__ x,
                                                    float* __restrict__ stats,
                                                    float* __restrict__ maskf) {
    __shared__ float ssum[256], ssq[256], scnt[256];
    float sum = 0.f, sq = 0.f, cnt = 0.f;
    for (int i = threadIdx.x; i < M_TOT; i += 256) {
        float v = x[i];
        float pad = (v == -1.0f) ? 1.f : 0.f;
        maskf[i] = pad;
        float vm = 1.f - pad;
        sum += v * vm; sq += v * v * vm; cnt += vm;
    }
    ssum[threadIdx.x] = sum; ssq[threadIdx.x] = sq; scnt[threadIdx.x] = cnt;
    __syncthreads();
    for (int o = 128; o > 0; o >>= 1) {
        if ((int)threadIdx.x < o) {
            ssum[threadIdx.x] += ssum[threadIdx.x + o];
            ssq[threadIdx.x]  += ssq[threadIdx.x + o];
            scnt[threadIdx.x] += scnt[threadIdx.x + o];
        }
        __syncthreads();
    }
    if (threadIdx.x == 0) {
        float c = scnt[0];
        float mean = ssum[0] / fmaxf(c, 1.f);
        float var = (ssq[0] - 2.f * mean * ssum[0] + c * mean * mean) / fmaxf(c - 1.f, 1.f);
        float stdv = sqrtf(var) + 1e-8f;
        stats[0] = clampf(mean, -100.f, 100.f);
        stats[1] = clampf(stdv, 0.01f, 100.f);
    }
}

// ---------------- embedding: rank-1 proj + gelu + pos ----------------
__global__ __launch_bounds__(256) void embed_kernel(const float* __restrict__ x,
                                                    const float* __restrict__ w_in,
                                                    const float* __restrict__ b_in,
                                                    const float* __restrict__ pos,
                                                    const float* __restrict__ stats,
                                                    float* __restrict__ h) {
    int idx = blockIdx.x * 256 + threadIdx.x;
    if (idx >= M_TOT * D_) return;
    int m = idx >> 7, d = idx & 127;
    float xv = x[m];
    bool pad = (xv == -1.0f);
    float xn = pad ? 0.f : clampf((xv - stats[0]) / stats[1], -10.f, 10.f);
    float hv = xn * w_in[d] + b_in[d];
    hv *= 0.9999950000374997f;           // 1/sqrt(1+1e-5)
    hv = gelu_f(hv);
    hv = clampf(hv, -100.f, 100.f);
    hv += 0.1f * pos[(m % T_) * D_ + d];
    h[idx] = hv;
}

// ---------------- LayerNorm -> bf16 (wave per row) ----------------
__global__ __launch_bounds__(256) void ln_kernel(const float* __restrict__ h,
                                                 const float* __restrict__ g,
                                                 const float* __restrict__ b,
                                                 bf16* __restrict__ out,
                                                 float eps_add) {
    int wave = threadIdx.x >> 5, lane = threadIdx.x & 31;
    int row = blockIdx.x * 8 + wave;
    const float* p = h + (long)row * D_;
    float v[4];
    float s = 0.f;
    #pragma unroll
    for (int j = 0; j < 4; j++) { v[j] = p[lane + 32 * j]; s += v[j]; }
    #pragma unroll
    for (int m = 1; m < 32; m <<= 1) s += __shfl_xor(s, m, 32);
    float mean = s * (1.f / 128.f);
    float q = 0.f;
    #pragma unroll
    for (int j = 0; j < 4; j++) { float d = v[j] - mean; q += d * d; }
    #pragma unroll
    for (int m = 1; m < 32; m <<= 1) q += __shfl_xor(q, m, 32);
    float rstd = rsqrtf(q * (1.f / 128.f) + 1e-5f);
    bf16* o = out + (long)row * D_;
    #pragma unroll
    for (int j = 0; j < 4; j++) {
        int d = lane + 32 * j;
        o[d] = (bf16)((v[j] - mean) * rstd * g[d] + b[d] + eps_add);
    }
}

// ---------------- weight packing (TRANSPOSED: out[n][k] = w[k][n]) ----------------
__global__ __launch_bounds__(256) void pack_qkvT_kernel(const float* __restrict__ wq,
                                                        const float* __restrict__ wk,
                                                        const float* __restrict__ wv,
                                                        const float* __restrict__ bq,
                                                        const float* __restrict__ bk,
                                                        const float* __restrict__ bv,
                                                        bf16* __restrict__ wqkvT,
                                                        float* __restrict__ bqkv) {
    int idx = blockIdx.x * 256 + threadIdx.x;
    if (idx < 384 * 128) {
        int j = idx / 128, k = idx % 128;     // output col j, contraction k
        int sel = j >> 7, jj = j & 127;
        const float* w = (sel == 0) ? wq : ((sel == 1) ? wk : wv);
        wqkvT[idx] = (bf16)w[k * 128 + jj];
    }
    if (idx < 384) {
        int sel = idx >> 7, jj = idx & 127;
        const float* bb = (sel == 0) ? bq : ((sel == 1) ? bk : bv);
        bqkv[idx] = bb[jj];
    }
}

__global__ __launch_bounds__(256) void f2bfT_kernel(const float* __restrict__ src,
                                                    bf16* __restrict__ dst, int K, int N) {
    int idx = blockIdx.x * 256 + threadIdx.x;
    if (idx >= K * N) return;
    int n = idx / K, k = idx % K;
    dst[idx] = (bf16)src[k * N + n];
}

// ---------------- V transpose: qkv[...,256+h*16+dk] -> vT[b,h,dk,TPAD] ----------------
__global__ __launch_bounds__(256) void vtrans_kernel(const bf16* __restrict__ qkv,
                                                     bf16* __restrict__ vT) {
    long idx = (long)blockIdx.x * 256 + threadIdx.x;
    long total = (long)B_ * H_ * DK_ * TPAD;
    if (idx >= total) return;
    int t = (int)(idx % TPAD);
    long r = idx / TPAD;
    int dk = (int)(r % DK_);
    int hh = (int)((r / DK_) % H_);
    int b  = (int)(r / (DK_ * H_));
    bf16 val = (bf16)0.f;
    if (t < T_) val = qkv[((long)(b * T_ + t)) * 384 + 256 + hh * 16 + dk];
    vT[idx] = val;
}

// ---------------- WMMA GEMM, 64x64 tile, B panel staged via Tensor Data Mover ----------
// BwT is the TRANSPOSED weight [N][K] row-major; the 64-column panel for this block is
// a contiguous 64*K-element region, fetched by ONE tensor_load_to_lds into LDS base 0.
// mode 0: QKV  -> out_bf = clip(acc+bias, +-10)
// mode 1: WO   -> h += tanh(*scal) * clip(acc+bias, +-100)
// mode 2: FFN1 -> out_bf = gelu(acc+bias)
// mode 3: FFN2 -> h = clip(h + tanh(*scal) * clip(acc+bias,+-100), +-1000)
__global__ __launch_bounds__(128) void gemm_kernel(const bf16* __restrict__ A,
                                                   const bf16* __restrict__ BwT,
                                                   const float* __restrict__ bias,
                                                   int K, int N, int mode,
                                                   bf16* __restrict__ out_bf,
                                                   float* __restrict__ h,
                                                   const float* __restrict__ scal) {
    extern __shared__ char smem[];
    bf16* ldsB = (bf16*)smem;                              // [64 cols][K], offset 0
    bf16* ldsA = (bf16*)(smem + (size_t)64 * K * 2);       // [64 rows][LDSS]
    int tid = threadIdx.x;
    int wave = tid >> 5, lane = tid & 31;
    int half = lane >> 4, ln = lane & 15;
    int m0 = blockIdx.y * 64, n0 = blockIdx.x * 64;

    // --- TDM: DMA the whole 64xK B panel into LDS (wave 0 issues; D# per ISA ch.8) ---
    if (tid < 32) {
        unsigned long long ga = (unsigned long long)(const void*)(BwT + (size_t)n0 * K);
        unsigned ne = (unsigned)(64 * K);                  // bf16 elements in panel
        u32x4 g0 = { 1u,                                   // count=1, user desc
                     0u,                                   // lds_addr = 0 (panel base)
                     (unsigned)(ga & 0xFFFFFFFFu),
                     (unsigned)(((ga >> 32) & 0x01FFFFFFu) | 0x80000000u) }; // addr[56:32] | type=2
        i32x8 g1 = { (int)0x00010000,                      // data_size=1 (2 bytes), no mask/flags
                     (int)((ne & 0xFFFFu) << 16),          // tensor_dim0[15:0] @ bits63:48
                     (int)(ne >> 16),                      // tensor_dim0[31:16]; tensor_dim1=0
                     (int)((ne & 0xFFFFu) << 16),          // tile_dim0 @ bits127:112
                     0, 0, 0, 0 };                         // tile_dim1/2=0, strides unused (1-D)
        i32x4 gz = { 0, 0, 0, 0 };
        i32x8 gz8 = { 0, 0, 0, 0, 0, 0, 0, 0 };
        __builtin_amdgcn_tensor_load_to_lds(g0, g1, gz, gz, gz8, 0);
        __builtin_amdgcn_s_wait_tensorcnt(0);
    }
    __syncthreads();

    v8f zero = {0.f, 0.f, 0.f, 0.f, 0.f, 0.f, 0.f, 0.f};
    v8f acc[4] = {zero, zero, zero, zero};

    int arow = tid >> 1, aseg = tid & 1;       // A loader mapping

    for (int k0 = 0; k0 < K; k0 += 32) {
        // stage A tile 64x32 (row-major)
        {
            const uint4* src = (const uint4*)(A + (long)(m0 + arow) * K + k0 + aseg * 16);
            uint4* dst = (uint4*)(ldsA + arow * LDSS + aseg * 16);
            dst[0] = src[0]; dst[1] = src[1];
        }
        if (k0 + 32 < K)
            __builtin_prefetch(A + (long)(m0 + arow) * K + k0 + 32, 0, 1);
        __syncthreads();

        Frag af;
        {
            const uint4* p0 = (const uint4*)(ldsA + (wave * 16 + ln) * LDSS + half * 8);
            const uint4* p1 = (const uint4*)(ldsA + (wave * 16 + ln) * LDSS + 16 + half * 8);
            af.u[0] = p0[0]; af.u[1] = p1[0];
        }
        #pragma unroll
        for (int nt = 0; nt < 4; nt++) {
            Frag bf_;
            const uint4* p = (const uint4*)(ldsB + (size_t)(nt * 16 + ln) * K + k0 + half * 8);
            bf_.u[0] = p[0]; bf_.u[1] = p[2];
            acc[nt] = __builtin_amdgcn_wmma_f32_16x16x32_bf16(
                false, af.v, false, bf_.v, (short)0, acc[nt], false, false);
        }
        __syncthreads();
    }

    float ta = scal ? tanhf(scal[0]) : 0.f;
    #pragma unroll
    for (int nt = 0; nt < 4; nt++) {
        int n = n0 + nt * 16 + ln;
        float bi = bias[n];
        #pragma unroll
        for (int v = 0; v < 8; v++) {
            int m = m0 + wave * 16 + half * 8 + v;
            float val = acc[nt][v] + bi;
            if (mode == 0) {
                out_bf[(long)m * N + n] = (bf16)clampf(val, -10.f, 10.f);
            } else if (mode == 2) {
                out_bf[(long)m * N + n] = (bf16)gelu_f(val);
            } else if (mode == 1) {
                float a = clampf(val, -100.f, 100.f);
                h[(long)m * N + n] += ta * a;
            } else {
                float f = clampf(val, -100.f, 100.f);
                h[(long)m * N + n] = clampf(h[(long)m * N + n] + ta * f, -1000.f, 1000.f);
            }
        }
    }
}

// ---------------- flash attention: one wave per 16-query tile per (b,h) ----------------
__global__ __launch_bounds__(128) void attn_kernel(const bf16* __restrict__ qkv,
                                                   const bf16* __restrict__ vT,
                                                   const float* __restrict__ maskf,
                                                   bf16* __restrict__ attn_out) {
    __shared__ bf16 ldsP[4 * 16 * LDSS];
    int wave = threadIdx.x >> 5, lane = threadIdx.x & 31;
    int qt = blockIdx.x * 4 + wave;
    if (qt >= NQT) return;
    int hh = blockIdx.y, b = blockIdx.z;
    int half = lane >> 4, ln = lane & 15;
    bf16* Pl = ldsP + wave * 16 * LDSS;

    // Q fragment (K-dim padded 16->32 with zeros)
    Frag qf;
    {
        int t = qt * 16 + ln; if (t >= T_) t = T_ - 1;
        const uint4* p = (const uint4*)(qkv + ((long)(b * T_ + t)) * 384 + hh * 16);
        qf.u[0] = half ? make_uint4(0u, 0u, 0u, 0u) : p[0];
        qf.u[1] = half ? make_uint4(0u, 0u, 0u, 0u) : p[1];
    }

    v8f zero = {0.f, 0.f, 0.f, 0.f, 0.f, 0.f, 0.f, 0.f};
    v8f acc = zero;
    float m_run[8], l_run[8];
    #pragma unroll
    for (int v = 0; v < 8; v++) { m_run[v] = -1e30f; l_run[v] = 0.f; }

    for (int kt = 0; kt < NKT; kt++) {
        int k0 = kt * 32;
        Frag kf0, kf1;
        if (half == 0) {
            int key0 = k0 + ln;      int kc0 = (key0 < T_) ? key0 : (T_ - 1);
            int key1 = k0 + 16 + ln; int kc1 = (key1 < T_) ? key1 : (T_ - 1);
            const uint4* p0 = (const uint4*)(qkv + ((long)(b * T_ + kc0)) * 384 + 128 + hh * 16);
            const uint4* p1 = (const uint4*)(qkv + ((long)(b * T_ + kc1)) * 384 + 128 + hh * 16);
            kf0.u[0] = p0[0]; kf0.u[1] = p0[1];
            kf1.u[0] = p1[0]; kf1.u[1] = p1[1];
        } else {
            kf0.u[0] = kf0.u[1] = make_uint4(0u, 0u, 0u, 0u);
            kf1.u[0] = kf1.u[1] = make_uint4(0u, 0u, 0u, 0u);
        }
        v8f s0 = __builtin_amdgcn_wmma_f32_16x16x32_bf16(false, qf.v, false, kf0.v, (short)0, zero, false, false);
        v8f s1 = __builtin_amdgcn_wmma_f32_16x16x32_bf16(false, qf.v, false, kf1.v, (short)0, zero, false, false);

        int key0 = k0 + ln, key1 = k0 + 16 + ln;
        float mk0 = (key0 < T_) ? maskf[b * T_ + key0] : 1.f;
        float mk1 = (key1 < T_) ? maskf[b * T_ + key1] : 1.f;
        #pragma unroll
        for (int v = 0; v < 8; v++) {
            float a0 = clampf(s0[v] * 0.25f, -50.f, 50.f);
            float a1 = clampf(s1[v] * 0.25f, -50.f, 50.f);
            if (mk0 != 0.f) a0 = -10000.f;
            if (mk1 != 0.f) a1 = -10000.f;
            s0[v] = a0; s1[v] = a1;
        }
        // online softmax per row (reduce across the 16 lanes of each half-wave)
        #pragma unroll
        for (int v = 0; v < 8; v++) {
            float mx = fmaxf(s0[v], s1[v]);
            #pragma unroll
            for (int msk = 1; msk < 16; msk <<= 1) mx = fmaxf(mx, __shfl_xor(mx, msk, 32));
            float nm = fmaxf(m_run[v], mx);
            float corr = __expf(m_run[v] - nm);
            float p0 = __expf(s0[v] - nm);
            float p1 = __expf(s1[v] - nm);
            float rs = p0 + p1;
            #pragma unroll
            for (int msk = 1; msk < 16; msk <<= 1) rs += __shfl_xor(rs, msk, 32);
            l_run[v] = l_run[v] * corr + rs;
            m_run[v] = nm;
            acc[v] *= corr;
            Pl[(half * 8 + v) * LDSS + ln]      = (bf16)p0;
            Pl[(half * 8 + v) * LDSS + 16 + ln] = (bf16)p1;
        }
        asm volatile("s_wait_dscnt 0" ::: "memory");   // P bounce ordering (single-wave)
        Frag pf;
        {
            const uint4* p0 = (const uint4*)(Pl + ln * LDSS + half * 8);
            const uint4* p1 = (const uint4*)(Pl + ln * LDSS + 16 + half * 8);
            pf.u[0] = p0[0]; pf.u[1] = p1[0];
        }
        Frag vf;
        {
            const uint4* p = (const uint4*)(vT + ((long)(b * H_ + hh) * DK_ + ln) * TPAD + k0 + half * 8);
            vf.u[0] = p[0]; vf.u[1] = p[2];
        }
        acc = __builtin_amdgcn_wmma_f32_16x16x32_bf16(false, pf.v, false, vf.v, (short)0, acc, false, false);
    }

    #pragma unroll
    for (int v = 0; v < 8; v++) {
        int t = qt * 16 + half * 8 + v;
        if (t < T_) {
            float o = acc[v] / (l_run[v] + 1e-10f);
            attn_out[((long)(b * T_ + t)) * D_ + hh * 16 + ln] = (bf16)o;
        }
    }
}

// ---------------- pooled classifier head ----------------
__global__ __launch_bounds__(128) void head_kernel(const float* __restrict__ h,
                                                   const float* __restrict__ maskf,
                                                   const float* __restrict__ lnc_g,
                                                   const float* __restrict__ lnc_b,
                                                   const float* __restrict__ wc1,
                                                   const float* __restrict__ bc1,
                                                   const float* __restrict__ wc2,
                                                   const float* __restrict__ bc2,
                                                   float* __restrict__ outp) {
    __shared__ float red[128];
    __shared__ float sh[128];
    __shared__ float zsh[64];
    int b = blockIdx.x, d = threadIdx.x;
    float sum = 0.f, cnt = 0.f;
    for (int t = 0; t < T_; t++) {
        float vm = 1.f - maskf[b * T_ + t];
        sum += h[((long)(b * T_ + t)) * D_ + d] * vm;
        cnt += vm;
    }
    float pooled = clampf(sum / fmaxf(cnt, 1.f), -100.f, 100.f);
    red[d] = pooled; __syncthreads();
    for (int o = 64; o > 0; o >>= 1) { if (d < o) red[d] += red[d + o]; __syncthreads(); }
    float mean = red[0] / 128.f; __syncthreads();
    float df = pooled - mean;
    red[d] = df * df; __syncthreads();
    for (int o = 64; o > 0; o >>= 1) { if (d < o) red[d] += red[d + o]; __syncthreads(); }
    float rstd = rsqrtf(red[0] / 128.f + 1e-5f);
    sh[d] = df * rstd * lnc_g[d] + lnc_b[d];
    __syncthreads();
    if (d < 64) {
        float a = bc1[d];
        for (int k = 0; k < 128; k++) a += sh[k] * wc1[k * 64 + d];
        zsh[d] = gelu_f(a);
    }
    __syncthreads();
    if (d < 2) {
        float a = bc2[d];
        for (int k = 0; k < 64; k++) a += zsh[k] * wc2[k * 2 + d];
        outp[b * 2 + d] = clampf(a, -20.f, 20.f);
    }
}

// ---------------- host-side orchestration ----------------
extern "C" void kernel_launch(void* const* d_in, const int* in_sizes, int n_in,
                              void* d_out, int out_size, void* d_ws, size_t ws_size,
                              hipStream_t stream) {
    (void)in_sizes; (void)n_in; (void)out_size; (void)ws_size;
    const float* x    = (const float*)d_in[0];
    const float* w_in = (const float*)d_in[1];
    const float* b_in = (const float*)d_in[2];
    const float* pos  = (const float*)d_in[3];
    const float* lnc_g = (const float*)d_in[58];
    const float* lnc_b = (const float*)d_in[59];
    const float* wc1   = (const float*)d_in[60];
    const float* bc1   = (const float*)d_in[61];
    const float* wc2   = (const float*)d_in[62];
    const float* bc2   = (const float*)d_in[63];

    char* ws = (char*)d_ws;
    size_t off = 0;
    auto alloc = [&](size_t bytes) -> void* {
        void* p = ws + off;
        off = (off + bytes + 255) & ~(size_t)255;
        return p;
    };
    float* stats  = (float*)alloc(2 * sizeof(float));
    float* maskf  = (float*)alloc((size_t)M_TOT * 4);
    float* h      = (float*)alloc((size_t)M_TOT * D_ * 4);
    bf16*  xln    = (bf16*)alloc((size_t)M_TOT * D_ * 2);
    bf16*  qkv    = (bf16*)alloc((size_t)M_TOT * 384 * 2);
    bf16*  vT     = (bf16*)alloc((size_t)B_ * H_ * DK_ * TPAD * 2);
    bf16*  attn_o = (bf16*)alloc((size_t)M_TOT * D_ * 2);
    bf16*  ffmid  = (bf16*)alloc((size_t)M_TOT * FF_ * 2);
    bf16*  wqkvT  = (bf16*)alloc(384 * 128 * 2);
    float* bqkv   = (float*)alloc(384 * 4);
    bf16*  woT    = (bf16*)alloc(128 * 128 * 2);
    bf16*  w1T    = (bf16*)alloc(512 * 128 * 2);
    bf16*  w2T    = (bf16*)alloc(128 * 512 * 2);

    stats_kernel<<<1, 256, 0, stream>>>(x, stats, maskf);
    embed_kernel<<<(M_TOT * D_ + 255) / 256, 256, 0, stream>>>(x, w_in, b_in, pos, stats, h);

    for (int l = 0; l < 3; l++) {
        int base = 4 + 18 * l;
        const float* wq   = (const float*)d_in[base + 0];
        const float* bq   = (const float*)d_in[base + 1];
        const float* wk   = (const float*)d_in[base + 2];
        const float* bk   = (const float*)d_in[base + 3];
        const float* wv   = (const float*)d_in[base + 4];
        const float* bv   = (const float*)d_in[base + 5];
        const float* wo   = (const float*)d_in[base + 6];
        const float* bo   = (const float*)d_in[base + 7];
        const float* ln1g = (const float*)d_in[base + 8];
        const float* ln1b = (const float*)d_in[base + 9];
        const float* ln2g = (const float*)d_in[base + 10];
        const float* ln2b = (const float*)d_in[base + 11];
        const float* w1   = (const float*)d_in[base + 12];
        const float* b1   = (const float*)d_in[base + 13];
        const float* w2   = (const float*)d_in[base + 14];
        const float* b2   = (const float*)d_in[base + 15];
        const float* alpha = (const float*)d_in[base + 16];
        const float* beta  = (const float*)d_in[base + 17];

        pack_qkvT_kernel<<<(384 * 128 + 255) / 256, 256, 0, stream>>>(wq, wk, wv, bq, bk, bv, wqkvT, bqkv);
        f2bfT_kernel<<<(16384 + 255) / 256, 256, 0, stream>>>(wo, woT, 128, 128);
        f2bfT_kernel<<<(65536 + 255) / 256, 256, 0, stream>>>(w1, w1T, 128, 512);
        f2bfT_kernel<<<(65536 + 255) / 256, 256, 0, stream>>>(w2, w2T, 512, 128);

        size_t sh128 = (size_t)64 * 128 * 2 + (size_t)64 * LDSS * 2;  // K=128 panels
        size_t sh512 = (size_t)64 * 512 * 2 + (size_t)64 * LDSS * 2;  // K=512 panel

        ln_kernel<<<M_TOT / 8, 256, 0, stream>>>(h, ln1g, ln1b, xln, 1e-8f);
        dim3 g0(384 / 64, M_TOT / 64);
        gemm_kernel<<<g0, 128, sh128, stream>>>(xln, wqkvT, bqkv, 128, 384, 0, qkv, nullptr, nullptr);
        vtrans_kernel<<<((B_ * H_ * DK_ * TPAD) + 255) / 256, 256, 0, stream>>>(qkv, vT);
        dim3 ga((NQT + 3) / 4, H_, B_);
        attn_kernel<<<ga, 128, 0, stream>>>(qkv, vT, maskf, attn_o);
        dim3 g1(128 / 64, M_TOT / 64);
        gemm_kernel<<<g1, 128, sh128, stream>>>(attn_o, woT, bo, 128, 128, 1, nullptr, h, alpha);
        ln_kernel<<<M_TOT / 8, 256, 0, stream>>>(h, ln2g, ln2b, xln, 0.f);
        dim3 g2(512 / 64, M_TOT / 64);
        gemm_kernel<<<g2, 128, sh128, stream>>>(xln, w1T, b1, 128, 512, 2, ffmid, nullptr, nullptr);
        dim3 g3(128 / 64, M_TOT / 64);
        gemm_kernel<<<g3, 128, sh512, stream>>>(ffmid, w2T, b2, 512, 128, 3, nullptr, h, beta);
    }

    head_kernel<<<B_, 128, 0, stream>>>(h, maskf, lnc_g, lnc_b, wc1, bc1, wc2, bc2, (float*)d_out);
}
